// FeaturePropogate_45964740001826
// MI455X (gfx1250) — matrix operand order repaired and checked by reading, not compiled, not run on hardware
//
#include <hip/hip_runtime.h>
#include <hip/hip_bf16.h>
#include <cstdint>
#include <cstddef>

typedef __bf16 bf16;
typedef __attribute__((ext_vector_type(16))) __bf16 v16bf;
typedef __attribute__((ext_vector_type(8)))  __bf16 v8bf;
typedef __attribute__((ext_vector_type(8)))  float  v8f;

// ------------------------------------------------------------------
// prep kernels
// ------------------------------------------------------------------
__global__ void cvt_f32_bf16(const float* __restrict__ src, bf16* __restrict__ dst, int n) {
  int i = blockIdx.x * blockDim.x + threadIdx.x;
  if (i < n) dst[i] = (bf16)src[i];
}

// alpha = g / sqrt(1 + 1e-5);  beta = alpha*b + be.  ab[0..n)=alpha, ab[n..2n)=beta
__global__ void bn_fold(const float* __restrict__ b, const float* __restrict__ g,
                        const float* __restrict__ be, float* __restrict__ ab, int n) {
  int i = blockIdx.x * blockDim.x + threadIdx.x;
  if (i < n) {
    float a = g[i] * rsqrtf(1.0f + 1e-5f);
    ab[i]     = a;
    ab[n + i] = a * b[i] + be[i];
  }
}

// ------------------------------------------------------------------
// top-8 smallest distances per point + inverse-distance weights.
// dist layout: [Bn][M][Npts] (the reference transposes, so scanning m is
// coalesced across adjacent threads which own adjacent n).
// ------------------------------------------------------------------
__global__ void knn_topk(const float* __restrict__ dist, int Bn, int M, int Npts,
                         int* __restrict__ idxOut, float* __restrict__ wOut) {
  int p = blockIdx.x * blockDim.x + threadIdx.x;
  if (p >= Bn * Npts) return;
  int b = p / Npts;
  int n = p - b * Npts;
  const float* dcol = dist + (size_t)b * M * Npts + n;

  float dk[8]; int ik[8];
#pragma unroll
  for (int j = 0; j < 8; ++j) { dk[j] = 3.402823466e38f; ik[j] = 0; }

  for (int m = 0; m < M; ++m) {
    float d = dcol[(size_t)m * Npts];
    if (d < dk[7]) {                       // strict < keeps earliest index on ties
      dk[7] = d; ik[7] = m;
#pragma unroll
      for (int s = 7; s > 0; --s) {
        if (dk[s] < dk[s - 1]) {
          float td = dk[s]; dk[s] = dk[s - 1]; dk[s - 1] = td;
          int   ti = ik[s]; ik[s] = ik[s - 1]; ik[s - 1] = ti;
        }
      }
    }
  }

  float w[8], wsum = 0.f;
#pragma unroll
  for (int j = 0; j < 8; ++j) { w[j] = 1.f / (dk[j] + 1e-8f); wsum += w[j]; }
  float inv = 1.f / (wsum + 1e-8f);
#pragma unroll
  for (int j = 0; j < 8; ++j) {
    idxOut[p * 8 + j] = ik[j];
    wOut[p * 8 + j]   = w[j] * inv;
  }
}

// ------------------------------------------------------------------
// X1[p][0..319] = concat( l1_cls_feat[b][n1][0..127],
//                         sum_k w*l2_cls_feat[b][idx][0..191] ), bf16
// ------------------------------------------------------------------
__global__ void build_x1(const float* __restrict__ l1feat, const float* __restrict__ l2feat,
                         const int* __restrict__ idx, const float* __restrict__ w,
                         bf16* __restrict__ X) {
  int p = blockIdx.x;                 // 0..8191 ; p = b*2048 + n1
  int b = p >> 11;
  int ik[8]; float wk[8];
#pragma unroll
  for (int j = 0; j < 8; ++j) { ik[j] = idx[p * 8 + j]; wk[j] = w[p * 8 + j]; }
  const float* f1 = l1feat + (size_t)p * 128;
  bf16* xr = X + (size_t)p * 320;
  for (int c = threadIdx.x; c < 320; c += blockDim.x) {
    float v;
    if (c < 128) {
      v = f1[c];
    } else {
      int cc = c - 128;
      v = 0.f;
#pragma unroll
      for (int j = 0; j < 8; ++j)
        v += wk[j] * l2feat[((size_t)b * 512 + ik[j]) * 192 + cc];
    }
    xr[c] = (bf16)v;
  }
}

// ------------------------------------------------------------------
// X0[p][0..703] = concat( radar[b][n0][0..63],
//                         sum_k w*F1[b*2048+idx][0..639] ), bf16
// ------------------------------------------------------------------
__global__ void build_x0(const float* __restrict__ radar, const bf16* __restrict__ F1,
                         const int* __restrict__ idx, const float* __restrict__ w,
                         bf16* __restrict__ X) {
  int p = blockIdx.x;                 // 0..32767 ; p = b*8192 + n0
  int b = p >> 13;
  int ik[8]; float wk[8];
#pragma unroll
  for (int j = 0; j < 8; ++j) { ik[j] = idx[p * 8 + j]; wk[j] = w[p * 8 + j]; }
  const float* rf = radar + (size_t)p * 64;
  bf16* xr = X + (size_t)p * 704;
  for (int c = threadIdx.x; c < 704; c += blockDim.x) {
    float v;
    if (c < 64) {
      v = rf[c];
    } else {
      int cc = c - 64;
      v = 0.f;
#pragma unroll
      for (int j = 0; j < 8; ++j)
        v += wk[j] * (float)F1[((size_t)b * 2048 + ik[j]) * 640 + cc];
    }
    xr[c] = (bf16)v;
  }
}

// ------------------------------------------------------------------
// WMMA GEMM + folded BN + ReLU.
//   A: [M][K] bf16 row-major (activations)     K % 32 == 0
//   W: [N][K] bf16 row-major (weights)         N % 128 == 0, M % 128 == 0
//   Y = relu(alpha[o] * (A . W^T)[m][o] + beta[o])
// Block tile 128(M) x 128(N), K-step 32, 8 waves (wave32) arranged 4(M) x 2(N);
// each wave owns a 32x64 strip -> 2 A-frags x 4 B-frags = 8
// v_wmma_f32_16x16x32_bf16 per K-step (1.5 ds_load_b128 per WMMA).
// FINAL=false: Y bf16 [M][N].  FINAL=true: Y fp32 in [B][N][8192] layout.
// ------------------------------------------------------------------
template <bool FINAL>
__global__ __launch_bounds__(256)
void gemm_bn_relu(const bf16* __restrict__ A, const bf16* __restrict__ W,
                  const float* __restrict__ alpha, const float* __restrict__ beta,
                  void* __restrict__ Yv, int M, int N, int K) {
  __shared__ bf16 As[128][32];   // 8 KB
  __shared__ bf16 Bs[128][32];   // 8 KB

  const int tid  = threadIdx.x;
  const int lane = tid & 31;
  const int wv   = tid >> 5;
  const int wm   = wv & 3;        // wave position along M (4 x 32 rows)
  const int wn   = wv >> 2;       // wave position along N (2 x 64 cols)
  const int mBase = blockIdx.y * 128;
  const int nBase = blockIdx.x * 128;

  v8f acc[2][4] = {};

  // staging: 4096 bf16 each for As and Bs -> 16 bf16 (2 x b128) per thread each
  const int sRow = tid >> 1, sOff = (tid & 1) * 16;
  const bf16* aSrc = A + (size_t)(mBase + sRow) * K + sOff;
  const bf16* bSrc = W + (size_t)(nBase + sRow) * K + sOff;

  for (int k0 = 0; k0 < K; k0 += 32) {
    *(uint4*)&As[sRow][sOff]     = *(const uint4*)(aSrc + k0);
    *(uint4*)&As[sRow][sOff + 8] = *(const uint4*)(aSrc + k0 + 8);
    *(uint4*)&Bs[sRow][sOff]     = *(const uint4*)(bSrc + k0);
    *(uint4*)&Bs[sRow][sOff + 8] = *(const uint4*)(bSrc + k0 + 8);
    if (k0 + 32 < K) __builtin_prefetch(aSrc + k0 + 32, 0, 1);  // global_prefetch_b8
    __syncthreads();

    // A fragments, 16-bit 16x32 layout (ISA 7.12.2):
    // lanes 0-15: row M=lane, K 0-7 (v0-3) & 16-23 (v4-7); lanes 16-31: K 8-15 & 24-31
    const int klo = (lane >> 4) * 8;
    v16bf afrag[2];
#pragma unroll
    for (int i = 0; i < 2; ++i) {
      const int am = wm * 32 + i * 16 + (lane & 15);
      v8bf alo = *(const v8bf*)&As[am][klo];
      v8bf ahi = *(const v8bf*)&As[am][klo + 16];
      afrag[i] = __builtin_shufflevector(alo, ahi,
          0, 1, 2, 3, 4, 5, 6, 7, 8, 9, 10, 11, 12, 13, 14, 15);
    }

    // B fragment: lane = column n (mod 16), same K split as A: lanes 0-15 K 0-7 & 16-23,
    // lanes 16-31 K 8-15 & 24-31 (weights stored [N][K] == B^T).
    const int koff = (lane >> 4) * 8;
#pragma unroll
    for (int s = 0; s < 4; ++s) {
      v8bf blo = *(const v8bf*)&Bs[wn * 64 + s * 16 + (lane & 15)][koff];
      v8bf bhi = *(const v8bf*)&Bs[wn * 64 + s * 16 + (lane & 15)][koff + 16];
      v16bf bfrag = __builtin_shufflevector(blo, bhi,
          0, 1, 2, 3, 4, 5, 6, 7, 8, 9, 10, 11, 12, 13, 14, 15);
#pragma unroll
      for (int i = 0; i < 2; ++i) {
        acc[i][s] = __builtin_amdgcn_wmma_f32_16x16x32_bf16(
            false, afrag[i], false, bfrag, (short)0, acc[i][s], false, false);
      }
    }
    __syncthreads();
  }

  // C/D layout: lanes 0-15 -> M = j, N = lane ; lanes 16-31 -> M = 8+j
#pragma unroll
  for (int i = 0; i < 2; ++i) {
    const int mRow0 = mBase + wm * 32 + i * 16 + (lane >> 4) * 8;
#pragma unroll
    for (int s = 0; s < 4; ++s) {
      const int o  = nBase + wn * 64 + s * 16 + (lane & 15);
      const float al = alpha[o];
      const float bt = beta[o];
#pragma unroll
      for (int j = 0; j < 8; ++j) {
        float v = al * acc[i][s][j] + bt;
        v = v > 0.f ? v : 0.f;
        const int m = mRow0 + j;
        if (FINAL) {
          const int b = m >> 13, n0 = m & 8191;             // m = b*8192 + n0
          ((float*)Yv)[((size_t)b * N + o) * 8192 + n0] = v;
        } else {
          ((bf16*)Yv)[(size_t)m * N + o] = (bf16)v;
        }
      }
    }
  }
}

// ------------------------------------------------------------------
// launch
// ------------------------------------------------------------------
extern "C" void kernel_launch(void* const* d_in, const int* in_sizes, int n_in,
                              void* d_out, int out_size, void* d_ws, size_t ws_size,
                              hipStream_t stream) {
  (void)in_sizes; (void)n_in; (void)out_size; (void)ws_size;

  const float* radar  = (const float*)d_in[0];   // [4,8192,64]
  const float* l1feat = (const float*)d_in[1];   // [4,2048,128]
  const float* l2feat = (const float*)d_in[2];   // [4,512,192]
  const float* l0to1  = (const float*)d_in[3];   // [4,2048,8192]
  const float* l1to2  = (const float*)d_in[4];   // [4,512,2048]
  const float* w21a = (const float*)d_in[5];
  const float* b21a = (const float*)d_in[6];
  const float* g21a = (const float*)d_in[7];
  const float* be21a = (const float*)d_in[8];
  const float* w21b = (const float*)d_in[9];
  const float* b21b = (const float*)d_in[10];
  const float* g21b = (const float*)d_in[11];
  const float* be21b = (const float*)d_in[12];
  const float* w10a = (const float*)d_in[13];
  const float* b10a = (const float*)d_in[14];
  const float* g10a = (const float*)d_in[15];
  const float* be10a = (const float*)d_in[16];
  const float* w10b = (const float*)d_in[17];
  const float* b10b = (const float*)d_in[18];
  const float* g10b = (const float*)d_in[19];
  const float* be10b = (const float*)d_in[20];

  uint8_t* ws = (uint8_t*)d_ws;
  size_t off = 0;
  auto carve = [&](size_t bytes) -> void* {
    off = (off + 255) & ~(size_t)255;
    void* p = ws + off;
    off += bytes;
    return p;
  };

  bf16* Wb21a = (bf16*)carve((size_t)640 * 320 * 2);
  bf16* Wb21b = (bf16*)carve((size_t)640 * 640 * 2);
  bf16* Wb10a = (bf16*)carve((size_t)640 * 704 * 2);
  bf16* Wb10b = (bf16*)carve((size_t)768 * 640 * 2);
  float* ab21a = (float*)carve((size_t)640 * 2 * 4);
  float* ab21b = (float*)carve((size_t)640 * 2 * 4);
  float* ab10a = (float*)carve((size_t)640 * 2 * 4);
  float* ab10b = (float*)carve((size_t)768 * 2 * 4);
  int*   idx1 = (int*)carve((size_t)8192 * 8 * 4);
  float* wgt1 = (float*)carve((size_t)8192 * 8 * 4);
  int*   idx0 = (int*)carve((size_t)32768 * 8 * 4);
  float* wgt0 = (float*)carve((size_t)32768 * 8 * 4);
  bf16* X1 = (bf16*)carve((size_t)8192 * 320 * 2);
  bf16* H1 = (bf16*)carve((size_t)8192 * 640 * 2);
  bf16* F1 = (bf16*)carve((size_t)8192 * 640 * 2);
  bf16* X0 = (bf16*)carve((size_t)32768 * 704 * 2);
  bf16* H0 = (bf16*)carve((size_t)32768 * 640 * 2);

  // --- weight conversion + BN fold ---
  {
    int n;
    n = 640 * 320; cvt_f32_bf16<<<(n + 255) / 256, 256, 0, stream>>>(w21a, Wb21a, n);
    n = 640 * 640; cvt_f32_bf16<<<(n + 255) / 256, 256, 0, stream>>>(w21b, Wb21b, n);
    n = 640 * 704; cvt_f32_bf16<<<(n + 255) / 256, 256, 0, stream>>>(w10a, Wb10a, n);
    n = 768 * 640; cvt_f32_bf16<<<(n + 255) / 256, 256, 0, stream>>>(w10b, Wb10b, n);
    bn_fold<<<3, 256, 0, stream>>>(b21a, g21a, be21a, ab21a, 640);
    bn_fold<<<3, 256, 0, stream>>>(b21b, g21b, be21b, ab21b, 640);
    bn_fold<<<3, 256, 0, stream>>>(b10a, g10a, be10a, ab10a, 640);
    bn_fold<<<3, 256, 0, stream>>>(b10b, g10b, be10b, ab10b, 768);
  }

  // --- level 2 -> level 1 ---
  knn_topk<<<(4 * 2048) / 256, 256, 0, stream>>>(l1to2, 4, 512, 2048, idx1, wgt1);
  build_x1<<<8192, 64, 0, stream>>>(l1feat, l2feat, idx1, wgt1, X1);
  gemm_bn_relu<false><<<dim3(640 / 128, 8192 / 128), 256, 0, stream>>>(
      X1, Wb21a, ab21a, ab21a + 640, H1, 8192, 640, 320);
  gemm_bn_relu<false><<<dim3(640 / 128, 8192 / 128), 256, 0, stream>>>(
      H1, Wb21b, ab21b, ab21b + 640, F1, 8192, 640, 640);

  // --- level 1 -> level 0 ---
  knn_topk<<<(4 * 8192) / 256, 256, 0, stream>>>(l0to1, 4, 2048, 8192, idx0, wgt0);
  build_x0<<<32768, 128, 0, stream>>>(radar, F1, idx0, wgt0, X0);
  gemm_bn_relu<false><<<dim3(640 / 128, 32768 / 128), 256, 0, stream>>>(
      X0, Wb10a, ab10a, ab10a + 640, H0, 32768, 640, 704);
  gemm_bn_relu<true><<<dim3(768 / 128, 32768 / 128), 256, 0, stream>>>(
      H0, Wb10b, ab10b, ab10b + 768, d_out, 32768, 768, 640);
}
